// Tacotron2Model_82154134438291
// MI455X (gfx1250) — matrix operand order repaired and checked by reading, not compiled, not run on hardware
//
#include <hip/hip_runtime.h>
#include <hip/hip_bf16.h>

// ---- model dims ----
#define BB 16
#define SS 256
#define TT 800
#define DD 512
#define MM 80
#define HH 1024
#define NHH 4
#define HDD 128
#define EHH 256

typedef _Float16 f16;
typedef __attribute__((ext_vector_type(16))) _Float16 v16h;
typedef __attribute__((ext_vector_type(8)))  float    v8f;
typedef __attribute__((ext_vector_type(4)))  unsigned int u32x4;

static __device__ __forceinline__ float sigf(float x) { return 1.0f / (1.0f + __expf(-x)); }

// WMMA f32 = f16 x f16 + f32, 16x16x32
static __device__ __forceinline__ v8f wmma32(v16h a, v16h b, v8f c) {
    return __builtin_amdgcn_wmma_f32_16x16x32_f16(false, a, false, b, (short)0, c, false, false);
}

// A-fragment: 16 halves per lane = chunks at +0 and +16 (ISA A 16x32 layout)
static __device__ __forceinline__ v16h load_afrag(const f16* p) {
    union { u32x4 u[2]; v16h h; } x;
    x.u[0] = *(const u32x4*)(p);
    x.u[1] = *(const u32x4*)(p + 16);
    return x.h;
}
// B-fragment: 16 contiguous halves per lane from packed buffer
static __device__ __forceinline__ v16h load_bfrag(const f16* p) {
    union { u32x4 u[2]; v16h h; } x;
    x.u[0] = *(const u32x4*)(p);
    x.u[1] = *(const u32x4*)(p + 8);
    return x.h;
}

// ---------------- embedding: gather + f32->f16 ----------------
__global__ void k_embed(const int* text, const float* emb, f16* out, int n) {
    int i = blockIdx.x * blockDim.x + threadIdx.x;
    if (i >= n) return;
    int row = i >> 9, d = i & 511;
    out[i] = (f16)emb[(long long)text[row] * DD + d];
}

// ---------------- generic B-fragment packer (f32 src -> packed f16) ----------
// src element (k,n) at src[baseOff + zo*srcOuter + zi*srcInner + k*sK + n*sN]
// dst frag layout: frag f = ks*NT + nt; halves d[(f*32+lane)*16 + h],
//   n = nt*16 + lane%16 ; k = ks*32 + (lane>=16 ? 8:0) + (h<8 ? h : h+8)   (ISA B 32x16 layout)
__global__ void k_pack(const float* src, long long baseOff, long long sK, long long sN,
                       int K, int N, int NT, int nK32,
                       long long srcOuter, long long srcInner, int innerN,
                       f16* dst, long long dstBS, long long total) {
    long long tid = (long long)blockIdx.x * blockDim.x + threadIdx.x;
    if (tid >= total) return;
    long long fragsPerZ = (long long)nK32 * NT * 32;
    long long z = tid / fragsPerZ;
    long long r = tid % fragsPerZ;
    int lane = (int)(r & 31);
    long long frag = r >> 5;
    int nt = (int)(frag % NT);
    int ks = (int)(frag / NT);
    int zo = (int)(z / innerN), zi = (int)(z % innerN);
    const float* s = src + baseOff + (long long)zo * srcOuter + (long long)zi * srcInner;
    f16* d = dst + z * dstBS + (frag * 32 + lane) * 16;
    int n = nt * 16 + (lane & 15);
    int kb = ks * 32 + ((lane & 16) ? 8 : 0);
    for (int h = 0; h < 16; ++h) {
        int k = kb + (h < 8 ? h : h + 8);
        float v = (k < K && n < N) ? s[(long long)k * sK + (long long)n * sN] : 0.0f;
        d[h] = (f16)v;
    }
}

// ---------------- generic batched WMMA GEMM -------------------
// C[m,n] = sum_k A[m,k]*Bpacked[k,n] (+bias[n]); 8 waves x 4 n-tiles per block.
// All 4 WMMAs issued unconditionally (inactive tiles clamped to a valid tile,
// stores predicated) -> no exec manipulation around v_wmma.
__global__ void __launch_bounds__(256) k_gemm(
    const f16* A, int lda, long long aO, long long aI,
    const f16* Bp, long long bO, long long bI,
    const float* bias,
    float* Cf, f16* Ch, int ldc, long long cO, long long cI,
    int M, int N, int Kp, int NT, int innerN) {
    int z = blockIdx.z;
    int zo = z / innerN, zi = z % innerN;
    const f16* Az = A + (long long)zo * aO + (long long)zi * aI;
    const f16* Bz = Bp + (long long)zo * bO + (long long)zi * bI;
    long long cOff = (long long)zo * cO + (long long)zi * cI;
    int w = __builtin_amdgcn_readfirstlane(threadIdx.x >> 5);   // uniform wave id
    int lane = threadIdx.x & 31;
    int mt = blockIdx.y;
    int nt0 = blockIdx.x * 32 + w * 4;                          // uniform
    if (nt0 >= NT) return;                                      // scalar branch
    // clamp inactive tiles onto a valid fragment (redundant compute, no branch)
    int nt1 = nt0 + 1 < NT ? nt0 + 1 : NT - 1;
    int nt2 = nt0 + 2 < NT ? nt0 + 2 : NT - 1;
    int nt3 = nt0 + 3 < NT ? nt0 + 3 : NT - 1;
    v8f acc[4];
    for (int t = 0; t < 4; ++t) acc[t] = (v8f)0.0f;
    int m = mt * 16 + (lane & 15);
    const f16* arow = Az + (long long)m * lda + ((lane & 16) ? 8 : 0);
    int nks = Kp >> 5;
    for (int ks = 0; ks < nks; ++ks) {
        v16h a = load_afrag(arow + ks * 32);
        const f16* bb = Bz + (long long)ks * NT * 512 + lane * 16;
        v16h b0 = load_bfrag(bb + (long long)nt0 * 512);
        v16h b1 = load_bfrag(bb + (long long)nt1 * 512);
        v16h b2 = load_bfrag(bb + (long long)nt2 * 512);
        v16h b3 = load_bfrag(bb + (long long)nt3 * 512);
        acc[0] = wmma32(a, b0, acc[0]);
        acc[1] = wmma32(a, b1, acc[1]);
        acc[2] = wmma32(a, b2, acc[2]);
        acc[3] = wmma32(a, b3, acc[3]);
    }
    int rofs = (lane & 16) ? 8 : 0;
    for (int t = 0; t < 4; ++t) {
        if (nt0 + t >= NT) break;                               // uniform
        int col = (nt0 + t) * 16 + (lane & 15);
        if (col >= N) continue;
        float bv = bias ? bias[col] : 0.0f;
        for (int r = 0; r < 8; ++r) {
            int row = mt * 16 + rofs + r;
            if (row >= M) continue;
            float v = acc[t][r] + bv;
            long long off = cOff + (long long)row * ldc + col;
            if (Cf) Cf[off] = v;
            if (Ch) Ch[off] = (f16)v;
        }
    }
}

// ---------------- conv1d(k=5,pad=2) + BN + ReLU, tap-wise WMMA ------------
__global__ void __launch_bounds__(256) k_conv(
    const f16* X, const f16* Bp,  // 5 packed taps, each 16*32*512 halves
    const float* cb, const float* gamma, const float* beta,
    const float* mean, const float* var, f16* Y) {
    const int NT = 32;
    int b = blockIdx.z, mt = blockIdx.y;
    int w = __builtin_amdgcn_readfirstlane(threadIdx.x >> 5);   // uniform
    int lane = threadIdx.x & 31;
    int nt0 = w * 4;
    v8f acc[4];
    for (int t = 0; t < 4; ++t) acc[t] = (v8f)0.0f;
    int sm = mt * 16 + (lane & 15);
    for (int tp = 0; tp < 5; ++tp) {
        int srow = sm + tp - 2;
        bool ok = (srow >= 0 && srow < SS);
        const f16* arow = X + ((long long)(b * SS) + srow) * DD + ((lane & 16) ? 8 : 0);
        const f16* tb = Bp + (long long)tp * (16 * 32 * 512);
        for (int ks = 0; ks < 16; ++ks) {
            v16h a = (v16h)(_Float16)0;
            if (ok) a = load_afrag(arow + ks * 32);             // lane-predicated load only
            const f16* bb = tb + ((long long)ks * NT + nt0) * 512 + lane * 16;
            { v16h bf = load_bfrag(bb);         acc[0] = wmma32(a, bf, acc[0]); }
            { v16h bf = load_bfrag(bb + 512);   acc[1] = wmma32(a, bf, acc[1]); }
            { v16h bf = load_bfrag(bb + 1024);  acc[2] = wmma32(a, bf, acc[2]); }
            { v16h bf = load_bfrag(bb + 1536);  acc[3] = wmma32(a, bf, acc[3]); }
        }
    }
    int rofs = (lane & 16) ? 8 : 0;
    for (int t = 0; t < 4; ++t) {
        int o = (nt0 + t) * 16 + (lane & 15);
        float sc = gamma[o] * rsqrtf(var[o] + 1e-5f);
        float sh = beta[o] - mean[o] * sc;
        float bc = cb[o];
        for (int r = 0; r < 8; ++r) {
            int srow = mt * 16 + rofs + r;
            float v = (acc[t][r] + bc) * sc + sh;
            v = v > 0.0f ? v : 0.0f;
            Y[((long long)(b * SS) + srow) * DD + o] = (f16)v;
        }
    }
}

// ---------------- encoder LSTM scan (persistent, 1 WG per direction) -------
// Gate-aligned waves: wave u owns units [u*16,(u+1)*16) for all 4 gates,
// so the c/h update is wave-local (no gate exchange). h lives in LDS (f16),
// c lives in registers. 16 waves x (1 A-frag + 4 B-frags + 4 wmma) x 8 ksteps.
__global__ void __launch_bounds__(512) k_scan(
    const f16* WhhF, const f16* WhhB,
    const float* GxF, const float* GxB, f16* enc) {
    __shared__ f16 hL[16 * EHH];
    int dir = blockIdx.x;
    const f16* Bp = dir ? WhhB : WhhF;
    const float* Gx = dir ? GxB : GxF;
    int w = __builtin_amdgcn_readfirstlane(threadIdx.x >> 5);   // uniform
    int lane = threadIdx.x & 31;
    for (int i = threadIdx.x; i < 16 * EHH; i += 512) hL[i] = (f16)0;
    __syncthreads();
    float c[8];
    for (int r = 0; r < 8; ++r) c[r] = 0.0f;
    int u = w;                       // unit tile 0..15
    int ln = lane & 15;
    int rofs = (lane & 16) ? 8 : 0;
    int j = u * 16 + ln;             // unit 0..255
    const f16* arow = &hL[ln * EHH + ((lane & 16) ? 8 : 0)];
    const int NT = 64;
    for (int step = 0; step < SS; ++step) {
        int s = dir ? (SS - 1 - step) : step;
        v8f ai, af, ag, ao;
        for (int r = 0; r < 8; ++r) {
            long long base = ((long long)(r + rofs) * SS + s) * 1024;
            ai[r] = Gx[base + j];
            af[r] = Gx[base + 256 + j];
            ag[r] = Gx[base + 512 + j];
            ao[r] = Gx[base + 768 + j];
        }
        for (int ks = 0; ks < 8; ++ks) {
            v16h a = load_afrag(arow + ks * 32);
            v16h b0 = load_bfrag(Bp + ((long long)ks * NT + u) * 512 + lane * 16);
            v16h b1 = load_bfrag(Bp + ((long long)ks * NT + 16 + u) * 512 + lane * 16);
            v16h b2 = load_bfrag(Bp + ((long long)ks * NT + 32 + u) * 512 + lane * 16);
            v16h b3 = load_bfrag(Bp + ((long long)ks * NT + 48 + u) * 512 + lane * 16);
            ai = wmma32(a, b0, ai);
            af = wmma32(a, b1, af);
            ag = wmma32(a, b2, ag);
            ao = wmma32(a, b3, ao);
        }
        __syncthreads();   // all reads of old h done
        for (int r = 0; r < 8; ++r) {
            float cc = sigf(af[r]) * c[r] + sigf(ai[r]) * tanhf(ag[r]);
            c[r] = cc;
            float hh = sigf(ao[r]) * tanhf(cc);
            int bidx = r + rofs;
            hL[bidx * EHH + j] = (f16)hh;
            enc[((long long)(bidx * SS) + s) * DD + dir * EHH + j] = (f16)hh;
        }
        __syncthreads();   // new h visible
    }
}

// ---------------- dec_in (teacher forcing, padded to 96 cols) ----------------
__global__ void k_decin(const float* mel_t, f16* out, int n) {
    int i = blockIdx.x * blockDim.x + threadIdx.x;
    if (i >= n) return;
    int row = i / 96, col = i % 96;
    int b = row / TT, t = row % TT;
    float v = 0.0f;
    if (col < MM && t > 0) v = mel_t[((long long)(b * TT) + (t - 1)) * MM + col];
    out[i] = (f16)v;
}

// ---------------- softmax over S=256 per row, one wave per row -------------
__global__ void __launch_bounds__(256) k_softmax(const float* S, f16* A, int rows) {
    int w = threadIdx.x >> 5, lane = threadIdx.x & 31;
    int row = blockIdx.x * 8 + w;
    if (row >= rows) return;
    const float* p = S + (long long)row * 256;
    const float scale = 0.088388347648318447f;  // 1/sqrt(128)
    float v[8], mx = -1e30f;
    for (int i = 0; i < 8; ++i) { v[i] = p[lane + 32 * i] * scale; mx = fmaxf(mx, v[i]); }
    for (int o = 16; o > 0; o >>= 1) mx = fmaxf(mx, __shfl_xor(mx, o, 32));
    float sum = 0.0f;
    for (int i = 0; i < 8; ++i) { v[i] = __expf(v[i] - mx); sum += v[i]; }
    for (int o = 16; o > 0; o >>= 1) sum += __shfl_xor(sum, o, 32);
    float inv = 1.0f / sum;
    f16* q = A + (long long)row * 256;
    for (int i = 0; i < 8; ++i) q[lane + 32 * i] = (f16)(v[i] * inv);
}

// ---------------- (B,NH,T,HD) f32 -> (B,T,D) f16 --------------------------
__global__ void k_ctx2h(const float* ctx, f16* out, int n) {
    int i = blockIdx.x * blockDim.x + threadIdx.x;
    if (i >= n) return;
    int col = i & 511, row = i >> 9;
    int h = col >> 7, d = col & 127;
    int b = row / TT, t = row % TT;
    out[i] = (f16)ctx[(((long long)(b * NHH + h) * TT) + t) * HDD + d];
}

// ---------------- xcat = [ctx_o(512) | dec_in(80) | pad(16)] f16 -----------
__global__ void k_xcat(const float* O, const f16* din, f16* out, int n) {
    int i = blockIdx.x * blockDim.x + threadIdx.x;
    if (i >= n) return;
    int row = i / 608, col = i % 608;
    f16 v;
    if (col < 512) v = (f16)O[(long long)row * 512 + col];
    else if (col < 592) v = din[(long long)row * 96 + (col - 512)];
    else v = (f16)0;
    out[i] = v;
}

// ---------------- fused GEMM + stateless LSTM cell (c0=0; f-gate dead) -----
// wave u owns units [u*16,(u+1)*16): n-tiles u (i), 128+u (g), 192+u (o)
__global__ void __launch_bounds__(256) k_cell(
    const f16* A, int lda, int Kp, const f16* Bp, const float* bias, f16* Hout) {
    const int NT = 256;
    int w = __builtin_amdgcn_readfirstlane(threadIdx.x >> 5);   // uniform
    int lane = threadIdx.x & 31;
    int mt = blockIdx.y;
    int u = blockIdx.x * 8 + w;   // 0..63, uniform
    v8f ai = (v8f)0.0f, ag = (v8f)0.0f, ao = (v8f)0.0f;
    int m = mt * 16 + (lane & 15);
    const f16* arow = A + (long long)m * lda + ((lane & 16) ? 8 : 0);
    int nks = Kp >> 5;
    for (int ks = 0; ks < nks; ++ks) {
        v16h a = load_afrag(arow + ks * 32);
        v16h bi = load_bfrag(Bp + ((long long)ks * NT + u) * 512 + lane * 16);
        v16h bg = load_bfrag(Bp + ((long long)ks * NT + 128 + u) * 512 + lane * 16);
        v16h bo = load_bfrag(Bp + ((long long)ks * NT + 192 + u) * 512 + lane * 16);
        ai = wmma32(a, bi, ai);
        ag = wmma32(a, bg, ag);
        ao = wmma32(a, bo, ao);
    }
    int j = u * 16 + (lane & 15);
    float b_i = bias[j], b_g = bias[2048 + j], b_o = bias[3072 + j];
    int rofs = (lane & 16) ? 8 : 0;
    for (int r = 0; r < 8; ++r) {
        int row = mt * 16 + rofs + r;
        float cc = sigf(ai[r] + b_i) * tanhf(ag[r] + b_g);
        float hh = sigf(ao[r] + b_o) * tanhf(cc);
        Hout[(long long)row * 1024 + j] = (f16)hh;
    }
}

// ---------------- stop logits: one wave per row dot(h2, stop_W) ------------
__global__ void __launch_bounds__(256) k_stop(const f16* H2, const float* sw,
                                              const float* sb, float* out) {
    int w = threadIdx.x >> 5, lane = threadIdx.x & 31;
    int row = blockIdx.x * 8 + w;
    if (row >= BB * TT) return;
    const f16* p = H2 + (long long)row * 1024;
    float s = 0.0f;
    for (int i = 0; i < 32; ++i) s += (float)p[lane + 32 * i] * sw[lane + 32 * i];
    for (int o = 16; o > 0; o >>= 1) s += __shfl_xor(s, o, 32);
    if (lane == 0) out[(long long)BB * TT * 80 + row] = s + sb[0];
}

extern "C" void kernel_launch(void* const* d_in, const int* in_sizes, int n_in,
                              void* d_out, int out_size, void* d_ws, size_t ws_size,
                              hipStream_t stream) {
    (void)in_sizes; (void)n_in; (void)out_size; (void)ws_size;
    const int*   text  = (const int*)d_in[0];
    const float* mel_t = (const float*)d_in[1];
    const float* emb   = (const float*)d_in[2];
    const float* convw = (const float*)d_in[3];
    const float* convb = (const float*)d_in[4];
    const float* bn_g  = (const float*)d_in[5];
    const float* bn_b  = (const float*)d_in[6];
    const float* bn_m  = (const float*)d_in[7];
    const float* bn_v  = (const float*)d_in[8];
    const float* WihF  = (const float*)d_in[9];
    const float* WhhF  = (const float*)d_in[10];
    const float* bF    = (const float*)d_in[11];
    const float* WihB  = (const float*)d_in[12];
    const float* WhhB  = (const float*)d_in[13];
    const float* bB    = (const float*)d_in[14];
    const float* Wq = (const float*)d_in[15]; const float* bq = (const float*)d_in[16];
    const float* Wk = (const float*)d_in[17]; const float* bk = (const float*)d_in[18];
    const float* Wv = (const float*)d_in[19]; const float* bv = (const float*)d_in[20];
    const float* Wo = (const float*)d_in[21]; const float* bo = (const float*)d_in[22];
    const float* W1 = (const float*)d_in[23]; const float* b1 = (const float*)d_in[24];
    const float* W2 = (const float*)d_in[25]; const float* b2 = (const float*)d_in[26];
    const float* melW = (const float*)d_in[27]; const float* melB = (const float*)d_in[28];
    const float* stopW = (const float*)d_in[29]; const float* stopB = (const float*)d_in[30];
    float* out = (float*)d_out;

    char* ws = (char*)d_ws;
    size_t cur = 0;
    auto alloc = [&](size_t bytes) -> char* {
        char* p = ws + cur; cur += (bytes + 255) & ~(size_t)255; return p;
    };

    f16* xEmb = (f16*)alloc(2097152ll * 2);
    f16* bufA = (f16*)alloc(2097152ll * 2);
    f16* bufB = (f16*)alloc(2097152ll * 2);
    f16* pConv = (f16*)alloc(15ll * 262144 * 2);
    f16* pWihF = (f16*)alloc(524288ll * 2);
    f16* pWihB = (f16*)alloc(524288ll * 2);
    f16* pWhhF = (f16*)alloc(262144ll * 2);
    f16* pWhhB = (f16*)alloc(262144ll * 2);
    f16* pWq = (f16*)alloc(49152ll * 2);
    f16* pWk = (f16*)alloc(262144ll * 2);
    f16* pWv = (f16*)alloc(262144ll * 2);
    f16* pWo = (f16*)alloc(262144ll * 2);
    f16* pW1 = (f16*)alloc(2490368ll * 2);
    f16* pW2 = (f16*)alloc(4194304ll * 2);
    f16* pMel = (f16*)alloc(81920ll * 2);
    float* GxF = (float*)alloc(4194304ll * 4);
    float* GxB = (float*)alloc(4194304ll * 4);
    f16* encH = (f16*)alloc(2097152ll * 2);
    f16* dIn  = (f16*)alloc(1228800ll * 2);
    f16* Qb   = (f16*)alloc(6553600ll * 2);
    float* Kb = (float*)alloc(2097152ll * 4);
    float* Vb = (float*)alloc(2097152ll * 4);
    f16* pScB = (f16*)alloc(2097152ll * 2);
    f16* pCxB = (f16*)alloc(2097152ll * 2);
    float* Sc = (float*)alloc(13107200ll * 4);
    f16* At   = (f16*)alloc(13107200ll * 2);
    float* Cx = (float*)alloc(6553600ll * 4);
    f16* CxH  = (f16*)alloc(6553600ll * 2);
    float* Ob = (float*)alloc(6553600ll * 4);
    f16* Xc   = (f16*)alloc(7782400ll * 2);
    f16* H1   = (f16*)alloc(13107200ll * 2);
    f16* H2   = (f16*)alloc(13107200ll * 2);

    auto pack = [&](const float* src, long long off, long long sK, long long sN,
                    int K, int N, f16* dst, int batch = 1,
                    long long so = 0, long long si = 0, int innerN = 1, long long dstBS = 0) {
        int Kp = (K + 31) & ~31, nK32 = Kp >> 5;
        int NT = ((N + 15) & ~15) >> 4;
        long long total = (long long)batch * nK32 * NT * 32;
        int blocks = (int)((total + 255) / 256);
        k_pack<<<blocks, 256, 0, stream>>>(src, off, sK, sN, K, N, NT, nK32,
                                           so, si, innerN, dst, dstBS, total);
    };
    auto gemm = [&](const f16* A, int lda, long long aO, long long aI,
                    const f16* Bp, long long bO, long long bI,
                    const float* bias, float* Cf, f16* Ch, int ldc,
                    long long cO, long long cI, int M, int N, int Kp,
                    int batch = 1, int innerN = 1) {
        int NT = ((N + 15) & ~15) >> 4;
        dim3 g((NT + 31) / 32, M / 16, batch);
        k_gemm<<<g, 256, 0, stream>>>(A, lda, aO, aI, Bp, bO, bI, bias,
                                      Cf, Ch, ldc, cO, cI, M, N, Kp, NT, innerN);
    };

    // ---- embedding ----
    k_embed<<<8192, 256, 0, stream>>>(text, emb, xEmb, 2097152);

    // ---- weight packs ----
    for (int L = 0; L < 3; ++L)
        for (int t = 0; t < 5; ++t)
            pack(convw, (long long)L * 512 * 512 * 5 + t, 5, 2560, 512, 512,
                 pConv + (long long)(L * 5 + t) * 262144);
    pack(WihF, 0, 1, 512, 512, 1024, pWihF);
    pack(WihB, 0, 1, 512, 512, 1024, pWihB);
    pack(WhhF, 0, 1, 256, 256, 1024, pWhhF);
    pack(WhhB, 0, 1, 256, 256, 1024, pWhhB);
    pack(Wq, 0, 1, 80, 80, 512, pWq);
    pack(Wk, 0, 1, 512, 512, 512, pWk);
    pack(Wv, 0, 1, 512, 512, 512, pWv);
    pack(Wo, 0, 1, 512, 512, 512, pWo);
    pack(W1, 0, 1, 592, 592, 4096, pW1);
    pack(W2, 0, 1, 1024, 1024, 4096, pW2);
    pack(melW, 0, 1, 1024, 1024, 80, pMel);

    // ---- conv stack (ping-pong) ----
    k_conv<<<dim3(1, 16, 16), 256, 0, stream>>>(xEmb, pConv, convb, bn_g, bn_b, bn_m, bn_v, bufA);
    k_conv<<<dim3(1, 16, 16), 256, 0, stream>>>(bufA, pConv + 5ll * 262144, convb + 512,
                                                bn_g + 512, bn_b + 512, bn_m + 512, bn_v + 512, bufB);
    k_conv<<<dim3(1, 16, 16), 256, 0, stream>>>(bufB, pConv + 10ll * 262144, convb + 1024,
                                                bn_g + 1024, bn_b + 1024, bn_m + 1024, bn_v + 1024, bufA);

    // ---- Gx = x @ Wih^T + b (hoisted out of the scan) ----
    gemm(bufA, 512, 0, 0, pWihF, 0, 0, bF, GxF, nullptr, 1024, 0, 0, 4096, 1024, 512);
    gemm(bufA, 512, 0, 0, pWihB, 0, 0, bB, GxB, nullptr, 1024, 0, 0, 4096, 1024, 512);

    // ---- bidirectional LSTM scan (2 persistent workgroups) ----
    k_scan<<<2, 512, 0, stream>>>(pWhhF, pWhhB, GxF, GxB, encH);

    // ---- attention ----
    k_decin<<<4800, 256, 0, stream>>>(mel_t, dIn, 1228800);
    gemm(dIn, 96, 0, 0, pWq, 0, 0, bq, nullptr, Qb, 512, 0, 0, 12800, 512, 96);
    gemm(encH, 512, 0, 0, pWk, 0, 0, bk, Kb, nullptr, 512, 0, 0, 4096, 512, 512);
    gemm(encH, 512, 0, 0, pWv, 0, 0, bv, Vb, nullptr, 512, 0, 0, 4096, 512, 512);
    // pack K^T and V per (b,h) into WMMA B-fragments
    pack(Kb, 0, 1, 512, 128, 256, pScB, 64, (long long)SS * 512, 128, 4, 32768);
    pack(Vb, 0, 512, 1, 256, 128, pCxB, 64, (long long)SS * 512, 128, 4, 32768);
    // scores[b,h] = Q_h (800x128) x K_h^T (128x256)
    gemm(Qb, 512, (long long)TT * 512, 128, pScB, 4ll * 32768, 32768, nullptr,
         Sc, nullptr, 256, 4ll * TT * 256, (long long)TT * 256, 800, 256, 128, 64, 4);
    k_softmax<<<6400, 256, 0, stream>>>(Sc, At, 51200);
    // ctx[b,h] = attn (800x256) x V_h (256x128)
    gemm(At, 256, 4ll * TT * 256, (long long)TT * 256, pCxB, 4ll * 32768, 32768, nullptr,
         Cx, nullptr, 128, 4ll * TT * 128, (long long)TT * 128, 800, 128, 256, 64, 4);
    k_ctx2h<<<25600, 256, 0, stream>>>(Cx, CxH, 6553600);
    gemm(CxH, 512, 0, 0, pWo, 0, 0, bo, Ob, nullptr, 512, 0, 0, 12800, 512, 512);

    // ---- decoder cells (fused GEMM + nonlinearity, no gate buffer) ----
    k_xcat<<<30400, 256, 0, stream>>>(Ob, dIn, Xc, 7782400);
    k_cell<<<dim3(8, 800), 256, 0, stream>>>(Xc, 608, 608, pW1, b1, H1);
    k_cell<<<dim3(8, 800), 256, 0, stream>>>(H1, 1024, 1024, pW2, b2, H2);

    // ---- outputs ----
    gemm(H2, 1024, 0, 0, pMel, 0, 0, melB, out, nullptr, 80, 0, 0, 12800, 80, 1024);
    k_stop<<<1600, 256, 0, stream>>>(H2, stopW, stopB, out);
}
